// SVPointNetEncoder_63007170232719
// MI455X (gfx1250) — matrix operand order repaired and checked by reading, not compiled, not run on hardware
//
#include <hip/hip_runtime.h>
#include <hip/hip_bf16.h>

#define SLOPE 0.2f
#define VN_EPS 1e-6f
#define KNN 20
#define BATCH 16
#define NPTS 2048
#define PTOT (BATCH * NPTS)
#define VOP 192   // vector-stream output channels padded 170 -> 192 (multiple of 64)

typedef __attribute__((ext_vector_type(16))) _Float16 v16h;
typedef __attribute__((ext_vector_type(8)))  _Float16 v8h;
typedef __attribute__((ext_vector_type(8)))  float    v8f;

__device__ __forceinline__ float lrelu(float x) { return x >= 0.f ? x : SLOPE * x; }

__device__ __forceinline__ void vn_act(float vl[3], const float dv[3]) {
  float dot = vl[0]*dv[0] + vl[1]*dv[1] + vl[2]*dv[2];
  float dn2 = dv[0]*dv[0] + dv[1]*dv[1] + dv[2]*dv[2] + VN_EPS;
  if (dot < 0.f) {
    float cc = (1.f - SLOPE) * dot / dn2;
    vl[0] -= cc * dv[0]; vl[1] -= cc * dv[1]; vl[2] -= cc * dv[2];
  }
}

// CDNA5 async global->LDS copy (ASYNCcnt-tracked). lds_off is the raw LDS
// byte offset (the kernel's single __shared__ block starts at offset 0).
__device__ __forceinline__ void async_ld_b128(unsigned lds_off, const void* gaddr) {
  asm volatile("global_load_async_to_lds_b128 %0, %1, off"
               :: "v"(lds_off), "v"((unsigned long long)(uintptr_t)gaddr)
               : "memory");
}
__device__ __forceinline__ void wait_async0() {
  asm volatile("s_wait_asynccnt 0" ::: "memory");
}

// ---------------------------------------------------------------- kNN ------
__global__ void knn_kernel(const float* __restrict__ x, int* __restrict__ idx) {
  int tid = threadIdx.x;
  int n = blockIdx.x * 256 + tid;
  int b = blockIdx.y;
  const float* xb = x + (size_t)b * 3 * NPTS;
  float cx = xb[n], cy = xb[NPTS + n], cz = xb[2 * NPTS + n];
  float bd[KNN]; int bi[KNN];
#pragma unroll
  for (int t = 0; t < KNN; ++t) { bd[t] = 3.4e38f; bi[t] = 0; }
  __shared__ float tx[256], ty[256], tz[256];
  for (int tile = 0; tile < NPTS; tile += 256) {
    __syncthreads();
    tx[tid] = xb[tile + tid];
    ty[tid] = xb[NPTS + tile + tid];
    tz[tid] = xb[2 * NPTS + tile + tid];
    __syncthreads();
    for (int jj = 0; jj < 256; ++jj) {
      float dx = tx[jj] - cx, dy = ty[jj] - cy, dz = tz[jj] - cz;
      float d = dx * dx + dy * dy + dz * dz;
      if (d < bd[KNN - 1]) {
        bd[KNN - 1] = d; bi[KNN - 1] = tile + jj;
#pragma unroll
        for (int t = KNN - 1; t > 0; --t) {
          if (bd[t] < bd[t - 1]) {
            float td = bd[t]; bd[t] = bd[t - 1]; bd[t - 1] = td;
            int ti = bi[t]; bi[t] = bi[t - 1]; bi[t - 1] = ti;
          }
        }
      }
    }
  }
  int* op = idx + (size_t)(b * NPTS + n) * KNN;
#pragma unroll
  for (int t = 0; t < KNN; ++t) op[t] = bi[t];
}

// ------------------------------------------- graph feature + conv_pos ------
__global__ void feat0_kernel(const float* __restrict__ x, const int* __restrict__ idx,
                             const float* __restrict__ Wv2s,
                             const float* __restrict__ Ws0, const float* __restrict__ bs0,
                             const float* __restrict__ Wv0, const float* __restrict__ Wd0,
                             float* __restrict__ s0, float* __restrict__ v0) {
  int pt = blockIdx.x * blockDim.x + threadIdx.x;
  if (pt >= PTOT) return;
  int b = pt >> 11, n = pt & (NPTS - 1);
  const float* xb = x + (size_t)b * 3 * NPTS;
  float cx = xb[n], cy = xb[NPTS + n], cz = xb[2 * NPTS + n];
  float accS[32], accV[10][3];
#pragma unroll
  for (int o = 0; o < 32; ++o) accS[o] = 0.f;
#pragma unroll
  for (int o = 0; o < 10; ++o) { accV[o][0] = 0.f; accV[o][1] = 0.f; accV[o][2] = 0.f; }
  const int* ip = idx + (size_t)pt * KNN;
  for (int kk = 0; kk < KNN; ++kk) {
    int j = ip[kk];
    float nx = xb[j], ny = xb[NPTS + j], nz = xb[2 * NPTS + j];
    float v[3][3];
    v[0][0] = nx - cx; v[0][1] = ny - cy; v[0][2] = nz - cz;
    v[1][0] = cx;      v[1][1] = cy;      v[1][2] = cz;
    v[2][0] = ny * cz - nz * cy; v[2][1] = nz * cx - nx * cz; v[2][2] = nx * cy - ny * cx;
    float dm[3][3];
#pragma unroll
    for (int m = 0; m < 3; ++m)
#pragma unroll
      for (int d = 0; d < 3; ++d)
        dm[m][d] = Wv2s[m * 3 + 0] * v[0][d] + Wv2s[m * 3 + 1] * v[1][d] + Wv2s[m * 3 + 2] * v[2][d];
    float s9[9];
#pragma unroll
    for (int c = 0; c < 3; ++c)
#pragma unroll
      for (int m = 0; m < 3; ++m)
        s9[c * 3 + m] = v[c][0] * dm[m][0] + v[c][1] * dm[m][1] + v[c][2] * dm[m][2];
#pragma unroll
    for (int o = 0; o < 32; ++o) {
      float a = bs0[o];
#pragma unroll
      for (int c = 0; c < 9; ++c) a += Ws0[o * 9 + c] * s9[c];
      accS[o] += lrelu(a);
    }
#pragma unroll
    for (int o = 0; o < 10; ++o) {
      float vl[3], dv[3];
#pragma unroll
      for (int d = 0; d < 3; ++d) {
        vl[d] = Wv0[o * 3 + 0] * v[0][d] + Wv0[o * 3 + 1] * v[1][d] + Wv0[o * 3 + 2] * v[2][d];
        dv[d] = Wd0[o * 3 + 0] * v[0][d] + Wd0[o * 3 + 1] * v[1][d] + Wd0[o * 3 + 2] * v[2][d];
      }
      vn_act(vl, dv);
      accV[o][0] += vl[0]; accV[o][1] += vl[1]; accV[o][2] += vl[2];
    }
  }
  const float inv = 1.f / KNN;
  float* sp = s0 + (size_t)pt * 32;
#pragma unroll
  for (int o = 0; o < 32; ++o) sp[o] = accS[o] * inv;
  float* vp = v0 + (size_t)pt * 30;
#pragma unroll
  for (int o = 0; o < 10; ++o) {
    vp[o * 3 + 0] = accV[o][0] * inv; vp[o * 3 + 1] = accV[o][1] * inv; vp[o * 3 + 2] = accV[o][2] * inv;
  }
}

// ------------------------------------------- conv1 + STN first block -------
__global__ void block1_kernel(const float* __restrict__ s0, const float* __restrict__ v0,
                              const float* __restrict__ Ws1, const float* __restrict__ bs1,
                              const float* __restrict__ Wv1, const float* __restrict__ Wd1,
                              const float* __restrict__ Wsf1, const float* __restrict__ bsf1,
                              const float* __restrict__ Wvf1, const float* __restrict__ Wdf1,
                              float* __restrict__ s1o, float* __restrict__ v1o,
                              float* __restrict__ sfw, float* __restrict__ vfw) {
  int pt = blockIdx.x * blockDim.x + threadIdx.x;
  if (pt >= PTOT) return;
  float s0r[32];
  const float* sp = s0 + (size_t)pt * 32;
#pragma unroll
  for (int c = 0; c < 32; ++c) s0r[c] = sp[c];
  float v0r[10][3];
  const float* vp = v0 + (size_t)pt * 30;
#pragma unroll
  for (int c = 0; c < 10; ++c) { v0r[c][0] = vp[c * 3 + 0]; v0r[c][1] = vp[c * 3 + 1]; v0r[c][2] = vp[c * 3 + 2]; }
  float s1r[32];
#pragma unroll
  for (int o = 0; o < 32; ++o) {
    float a = bs1[o];
#pragma unroll
    for (int c = 0; c < 32; ++c) a += Ws1[o * 32 + c] * s0r[c];
    s1r[o] = lrelu(a);
  }
  float v1r[10][3];
#pragma unroll
  for (int o = 0; o < 10; ++o) {
    float vl[3] = {0.f, 0.f, 0.f}, dv[3] = {0.f, 0.f, 0.f};
#pragma unroll
    for (int c = 0; c < 10; ++c)
#pragma unroll
      for (int d = 0; d < 3; ++d) {
        vl[d] += Wv1[o * 10 + c] * v0r[c][d];
        dv[d] += Wd1[o * 10 + c] * v0r[c][d];
      }
    vn_act(vl, dv);
    v1r[o][0] = vl[0]; v1r[o][1] = vl[1]; v1r[o][2] = vl[2];
  }
  float* so = s1o + (size_t)pt * 32;
  float* vo = v1o + (size_t)pt * 30;
#pragma unroll
  for (int o = 0; o < 32; ++o) so[o] = s1r[o];
#pragma unroll
  for (int o = 0; o < 10; ++o) { vo[o * 3 + 0] = v1r[o][0]; vo[o * 3 + 1] = v1r[o][1]; vo[o * 3 + 2] = v1r[o][2]; }
  float* fo = sfw + (size_t)pt * 64;
  for (int o = 0; o < 64; ++o) {
    float a = bsf1[o];
#pragma unroll
    for (int c = 0; c < 32; ++c) a += Wsf1[o * 32 + c] * s1r[c];
    fo[o] = lrelu(a);
  }
  float* fv = vfw + (size_t)pt * 63;
  for (int o = 0; o < 21; ++o) {
    float vl[3] = {0.f, 0.f, 0.f}, dv[3] = {0.f, 0.f, 0.f};
#pragma unroll
    for (int c = 0; c < 10; ++c)
#pragma unroll
      for (int d = 0; d < 3; ++d) {
        vl[d] += Wvf1[o * 10 + c] * v1r[c][d];
        dv[d] += Wdf1[o * 10 + c] * v1r[c][d];
      }
    vn_act(vl, dv);
    fv[o * 3 + 0] = vl[0]; fv[o * 3 + 1] = vl[1]; fv[o * 3 + 2] = vl[2];
  }
}

// --------------------------- generic mean over N (per batch) ---------------
template <typename T>
__global__ void mean_n_kernel(const T* __restrict__ X, float* __restrict__ out,
                              int rowStride, int R, int C, int outStride) {
  int b = blockIdx.x;
  int rc = blockIdx.y;
  int r = rc / C, c = rc % C;
  float sum = 0.f;
  for (int n = threadIdx.x; n < NPTS; n += 256)
    sum += (float)X[((size_t)(b * NPTS + n) * R + r) * rowStride + c];
  __shared__ float red[256];
  red[threadIdx.x] = sum;
  __syncthreads();
  for (int s = 128; s > 0; s >>= 1) {
    if (threadIdx.x < s) red[threadIdx.x] += red[threadIdx.x + s];
    __syncthreads();
  }
  if (threadIdx.x == 0) out[(size_t)(b * R + r) * outStride + c] = red[0] * (1.f / NPTS);
}

// ------------------------------------------- STN global block (fw2) --------
__global__ void fw2_kernel(const float* __restrict__ gsmean, const float* __restrict__ gvmean,
                           const float* __restrict__ Wsf2, const float* __restrict__ bsf2,
                           const float* __restrict__ Wvf2, const float* __restrict__ Wdf2,
                           float* __restrict__ gs2, float* __restrict__ gv2) {
  int b = blockIdx.x;
  int t = threadIdx.x;
  if (t < 32) {
    float a = bsf2[t];
    for (int c = 0; c < 64; ++c) a += Wsf2[t * 64 + c] * gsmean[b * 64 + c];
    gs2[b * 32 + t] = lrelu(a);
  } else if (t < 42) {
    int o = t - 32;
    float vl[3] = {0.f, 0.f, 0.f}, dv[3] = {0.f, 0.f, 0.f};
    for (int c = 0; c < 21; ++c) {
#pragma unroll
      for (int d = 0; d < 3; ++d) {
        float vv = gvmean[b * 63 + c * 3 + d];
        vl[d] += Wvf2[o * 21 + c] * vv;
        dv[d] += Wdf2[o * 21 + c] * vv;
      }
    }
    vn_act(vl, dv);
#pragma unroll
    for (int d = 0; d < 3; ++d) gv2[b * 30 + o * 3 + d] = vl[d];
  }
}

// ------------------------------- svcat + conv2; emits f16 GEMM inputs ------
__global__ void block2_kernel(const float* __restrict__ s1, const float* __restrict__ v1,
                              const float* __restrict__ gs2, const float* __restrict__ gv2,
                              const float* __restrict__ Ws2, const float* __restrict__ bs2,
                              const float* __restrict__ Wv2, const float* __restrict__ Wd2,
                              _Float16* __restrict__ s2h, _Float16* __restrict__ v2h) {
  int pt = blockIdx.x * blockDim.x + threadIdx.x;
  if (pt >= PTOT) return;
  int b = pt >> 11;
  float scat[64];
  const float* sp = s1 + (size_t)pt * 32;
#pragma unroll
  for (int c = 0; c < 32; ++c) scat[c] = sp[c];
#pragma unroll
  for (int c = 0; c < 32; ++c) scat[32 + c] = gs2[b * 32 + c];
  float vcat[20][3];
  const float* vp = v1 + (size_t)pt * 30;
#pragma unroll
  for (int c = 0; c < 10; ++c) {
    vcat[c][0] = vp[c * 3 + 0]; vcat[c][1] = vp[c * 3 + 1]; vcat[c][2] = vp[c * 3 + 2];
    vcat[10 + c][0] = gv2[b * 30 + c * 3 + 0];
    vcat[10 + c][1] = gv2[b * 30 + c * 3 + 1];
    vcat[10 + c][2] = gv2[b * 30 + c * 3 + 2];
  }
  _Float16* so = s2h + (size_t)pt * 64;
  for (int o = 0; o < 64; ++o) {
    float a = bs2[o];
#pragma unroll
    for (int c = 0; c < 64; ++c) a += Ws2[o * 64 + c] * scat[c];
    so[o] = (_Float16)lrelu(a);
  }
  for (int o = 0; o < 21; ++o) {
    float vl[3] = {0.f, 0.f, 0.f}, dv[3] = {0.f, 0.f, 0.f};
#pragma unroll
    for (int c = 0; c < 20; ++c)
#pragma unroll
      for (int d = 0; d < 3; ++d) {
        vl[d] += Wv2[o * 20 + c] * vcat[c][d];
        dv[d] += Wd2[o * 20 + c] * vcat[c][d];
      }
    vn_act(vl, dv);
#pragma unroll
    for (int d = 0; d < 3; ++d) v2h[((size_t)pt * 3 + d) * 32 + o] = (_Float16)vl[d];
  }
  for (int o = 21; o < 32; ++o)
#pragma unroll
    for (int d = 0; d < 3; ++d) v2h[((size_t)pt * 3 + d) * 32 + o] = (_Float16)0.f;
}

// ----------------------------- weight f32 -> f16 with zero padding ---------
__global__ void pack_w_kernel(const float* __restrict__ W, _Float16* __restrict__ Wh,
                              int O, int K, int Op, int Kp) {
  int i = blockIdx.x * blockDim.x + threadIdx.x;
  if (i >= Op * Kp) return;
  int k = i % Kp, o = i / Kp;
  Wh[i] = (o < O && k < K) ? (_Float16)W[o * K + k] : (_Float16)0.f;
}

// ----------------------------------- WMMA GEMM: C = act(A * W^T + b) -------
// Block tile 128(M) x 64(N) x 32(K), 256 threads = 8 waves (2 Mw x 4 Nw).
// Double-buffered LDS stages filled by global_load_async_to_lds_b128
// (ASYNCcnt): stage s+1's DMA overlaps stage s's WMMA work; one barrier per
// k-step. Rows padded to 80 B to break bank conflicts. Each wave holds 4
// accumulators (64 M-rows x 16 N-cols); all fragments are loaded before the
// 4 back-to-back v_wmma_f32_16x16x32_f16 so a single DScnt wait covers them.
// Requires M % 128 == 0, O % 64 == 0, Kp % 32 == 0.
#define GBM 128
#define GBN 64
#define LDSROW 80
#define LDSB0 (GBM * LDSROW)
#define STAGE ((GBM + GBN) * LDSROW)   // 15360 bytes per stage

template <int EPI>
__global__ void __launch_bounds__(256)
wmma_gemm_kernel(const _Float16* __restrict__ A, const _Float16* __restrict__ Wt,
                 const float* __restrict__ bias, _Float16* __restrict__ C,
                 int Kp, int O) {
  __shared__ char smem[2 * STAGE];
  const int tid = threadIdx.x;
  const int lane = tid & 31;
  const int wave = tid >> 5;
  const int mw = wave >> 2;    // 0..1
  const int nw = wave & 3;     // 0..3
  const size_t row0 = (size_t)blockIdx.x * GBM;
  const size_t col0 = (size_t)blockIdx.y * GBN;

  // staging assignment: 3 x 16B chunks per thread per k-step
  const int rA0 = tid >> 2,         cA0 = tid & 3;          // A chunks 0..255
  const int rA1 = (tid + 256) >> 2, cA1 = tid & 3;          // A chunks 256..511
  const int rB  = tid >> 2,         cB  = tid & 3;          // B chunks 0..255
  const unsigned ldsA0 = rA0 * LDSROW + cA0 * 16;
  const unsigned ldsA1 = rA1 * LDSROW + cA1 * 16;
  const unsigned ldsB  = LDSB0 + rB * LDSROW + cB * 16;
  const char* gA0 = (const char*)(A + (row0 + rA0) * Kp + cA0 * 8);
  const char* gA1 = (const char*)(A + (row0 + rA1) * Kp + cA1 * 8);
  const char* gB  = (const char*)(Wt + (col0 + rB) * Kp + cB * 8);

  // per-lane fragment source addresses in LDS (ISA 16-bit A/B layouts)
  const int fr  = lane & 15;
  const int ka0 = (lane >> 4) * 8;     // A: halfs {ka0..+7, ka0+16..+23}
  const int kb0 = (lane >> 4) * 8;     // B: halfs {kb0..+7, kb0+16..+23}
  const char* aBase = smem + (mw * 64 + fr) * LDSROW + ka0 * 2;
  const char* bBase = smem + LDSB0 + (nw * 16 + fr) * LDSROW + kb0 * 2;

  v8f acc[4] = {v8f{}, v8f{}, v8f{}, v8f{}};
  // prologue: fill stage 0
  async_ld_b128(ldsA0, gA0);
  async_ld_b128(ldsA1, gA1);
  async_ld_b128(ldsB,  gB);
  for (int kt = 0; kt < Kp; kt += 32) {
    const unsigned stg = ((unsigned)(kt >> 5) & 1u) * STAGE;
    wait_async0();
    __syncthreads();
    if (kt + 32 < Kp) {       // kick off next stage's DMA; overlaps the WMMAs
      const unsigned nstg = stg ^ STAGE;
      const size_t koff = (size_t)(kt + 32) * 2;
      async_ld_b128(ldsA0 + nstg, gA0 + koff);
      async_ld_b128(ldsA1 + nstg, gA1 + koff);
      async_ld_b128(ldsB  + nstg, gB  + koff);
    }
    // load all fragments first, then run the WMMAs back-to-back
    v8h b0 = *(const v8h*)(bBase + stg);
    v8h b1 = *(const v8h*)(bBase + stg + 32);
    v16h am[4];
#pragma unroll
    for (int t = 0; t < 4; ++t) {
      const char* ab = aBase + stg + t * 16 * LDSROW;
      v8h alo = *(const v8h*)(ab);
      v8h ahi = *(const v8h*)(ab + 32);
#pragma unroll
      for (int i = 0; i < 8; ++i) { am[t][i] = alo[i]; am[t][8 + i] = ahi[i]; }
    }
    v16h bm;
#pragma unroll
    for (int i = 0; i < 8; ++i) { bm[i] = b0[i]; bm[8 + i] = b1[i]; }
#pragma unroll
    for (int t = 0; t < 4; ++t)
      acc[t] = __builtin_amdgcn_wmma_f32_16x16x32_f16(false, am[t], false, bm, (short)0, acc[t],
                                                      false, false);
  }
  const int ncol = (int)col0 + nw * 16 + fr;
  const float bv = (EPI == 1) ? bias[ncol] : 0.f;
#pragma unroll
  for (int t = 0; t < 4; ++t) {
    const size_t mbase = row0 + mw * 64 + t * 16 + (lane >> 4) * 8;
#pragma unroll
    for (int j = 0; j < 8; ++j) {
      float xv = acc[t][j] + bv;
      if (EPI == 1) xv = lrelu(xv);
      C[(mbase + j) * O + ncol] = (_Float16)xv;
    }
  }
}

// -------------------------- VN leaky-relu over (x,y,z) row triples ---------
__global__ void vn_leaky_kernel(const _Float16* __restrict__ Vl, const _Float16* __restrict__ Dd,
                                _Float16* __restrict__ Vout, int O, int Op) {
  size_t i = (size_t)blockIdx.x * blockDim.x + threadIdx.x;
  if (i >= (size_t)PTOT * Op) return;
  int o = (int)(i % Op);
  size_t pt = i / Op;
  size_t r0 = pt * 3 * (size_t)Op + o;
  if (o >= O) {
    Vout[r0] = (_Float16)0.f; Vout[r0 + Op] = (_Float16)0.f; Vout[r0 + 2 * Op] = (_Float16)0.f;
    return;
  }
  float vl[3] = {(float)Vl[r0], (float)Vl[r0 + Op], (float)Vl[r0 + 2 * Op]};
  float dv[3] = {(float)Dd[r0], (float)Dd[r0 + Op], (float)Dd[r0 + 2 * Op]};
  vn_act(vl, dv);
  Vout[r0] = (_Float16)vl[0]; Vout[r0 + Op] = (_Float16)vl[1]; Vout[r0 + 2 * Op] = (_Float16)vl[2];
}

// ----------------------- build conv_fuse inputs (svcat with means) ---------
__global__ void pack_s4in_kernel(const _Float16* __restrict__ s3h, const float* __restrict__ smean,
                                 _Float16* __restrict__ s4in) {
  size_t i = (size_t)blockIdx.x * blockDim.x + threadIdx.x;
  if (i >= (size_t)PTOT * 1024) return;
  int c = (int)(i & 1023);
  int pt = (int)(i >> 10);
  int b = pt >> 11;
  s4in[i] = (c < 512) ? s3h[(size_t)pt * 512 + c] : (_Float16)smean[b * 512 + (c - 512)];
}

__global__ void pack_v4in_kernel(const _Float16* __restrict__ v3h, const float* __restrict__ vmean3,
                                 _Float16* __restrict__ v4in) {
  size_t i = (size_t)blockIdx.x * blockDim.x + threadIdx.x;
  if (i >= (size_t)PTOT * 3 * 352) return;
  int c = (int)(i % 352);
  size_t row = i / 352;                 // row = pt*3 + d
  int d = (int)(row % 3);
  int b = (int)((row / 3) >> 11);
  _Float16 val;
  if (c < 170)      val = v3h[row * VOP + c];
  else if (c < 340) val = (_Float16)vmean3[((size_t)(b * 3 + d)) * VOP + (c - 170)];
  else              val = (_Float16)0.f;
  v4in[i] = val;
}

// ----------------------------------- final SVFuse invariants ---------------
__global__ void fuse_kernel(const float* __restrict__ vm4, const float* __restrict__ Wfuse,
                            float* __restrict__ out) {
  int b = blockIdx.x;
  int t = threadIdx.x;
  __shared__ float dmat[9];
  if (t < 9) {
    int m = t / 3, dd = t % 3;
    float a = 0.f;
    for (int c = 0; c < 170; ++c) a += Wfuse[m * 170 + c] * vm4[((size_t)(b * 3 + dd)) * VOP + c];
    dmat[t] = a;
  }
  __syncthreads();
  if (t < 510) {
    int c = t / 3, m = t % 3;
    float s = 0.f;
#pragma unroll
    for (int dd = 0; dd < 3; ++dd) s += vm4[((size_t)(b * 3 + dd)) * VOP + c] * dmat[m * 3 + dd];
    out[(size_t)b * 1022 + 512 + c * 3 + m] = s;
  }
}

// ===========================================================================
extern "C" void kernel_launch(void* const* d_in, const int* in_sizes, int n_in,
                              void* d_out, int out_size, void* d_ws, size_t ws_size,
                              hipStream_t stream) {
  (void)in_sizes; (void)n_in; (void)out_size; (void)ws_size;
  const float* x    = (const float*)d_in[0];
  const float* Wv2s = (const float*)d_in[1];
  const float* Ws0 = (const float*)d_in[2];  const float* bs0 = (const float*)d_in[3];
  const float* Wv0 = (const float*)d_in[4];  const float* Wd0 = (const float*)d_in[5];
  const float* Ws1 = (const float*)d_in[6];  const float* bs1 = (const float*)d_in[7];
  const float* Wv1 = (const float*)d_in[8];  const float* Wd1 = (const float*)d_in[9];
  const float* Wsf1 = (const float*)d_in[10]; const float* bsf1 = (const float*)d_in[11];
  const float* Wvf1 = (const float*)d_in[12]; const float* Wdf1 = (const float*)d_in[13];
  const float* Wsf2 = (const float*)d_in[14]; const float* bsf2 = (const float*)d_in[15];
  const float* Wvf2 = (const float*)d_in[16]; const float* Wdf2 = (const float*)d_in[17];
  const float* Ws2 = (const float*)d_in[18]; const float* bs2 = (const float*)d_in[19];
  const float* Wv2 = (const float*)d_in[20]; const float* Wd2 = (const float*)d_in[21];
  const float* Ws3 = (const float*)d_in[22]; const float* bs3 = (const float*)d_in[23];
  const float* Wv3 = (const float*)d_in[24]; const float* Wd3 = (const float*)d_in[25];
  const float* Ws4 = (const float*)d_in[26]; const float* bs4 = (const float*)d_in[27];
  const float* Wv4 = (const float*)d_in[28]; const float* Wd4 = (const float*)d_in[29];
  const float* Wfuse = (const float*)d_in[30];
  float* out = (float*)d_out;

  char* base = (char*)d_ws;
  size_t off = 0;
  auto alloc = [&](size_t bytes) -> void* {
    void* p = base + off;
    off += (bytes + 255) & ~(size_t)255;
    return p;
  };
  int*      idx    = (int*)alloc((size_t)PTOT * KNN * 4);
  float*    s0     = (float*)alloc((size_t)PTOT * 32 * 4);
  float*    v0     = (float*)alloc((size_t)PTOT * 30 * 4);
  float*    s1     = (float*)alloc((size_t)PTOT * 32 * 4);
  float*    v1     = (float*)alloc((size_t)PTOT * 30 * 4);
  float*    sfw    = (float*)alloc((size_t)PTOT * 64 * 4);
  float*    vfw    = (float*)alloc((size_t)PTOT * 63 * 4);
  float*    gsmean = (float*)alloc((size_t)BATCH * 64 * 4);
  float*    gvmean = (float*)alloc((size_t)BATCH * 63 * 4);
  float*    gs2    = (float*)alloc((size_t)BATCH * 32 * 4);
  float*    gv2    = (float*)alloc((size_t)BATCH * 30 * 4);
  _Float16* s2h    = (_Float16*)alloc((size_t)PTOT * 64 * 2);
  _Float16* v2h    = (_Float16*)alloc((size_t)PTOT * 3 * 32 * 2);
  _Float16* Ws3h   = (_Float16*)alloc((size_t)512 * 64 * 2);
  _Float16* Wv3h   = (_Float16*)alloc((size_t)VOP * 32 * 2);
  _Float16* Wd3h   = (_Float16*)alloc((size_t)VOP * 32 * 2);
  _Float16* Ws4h   = (_Float16*)alloc((size_t)512 * 1024 * 2);
  _Float16* Wv4h   = (_Float16*)alloc((size_t)VOP * 352 * 2);
  _Float16* Wd4h   = (_Float16*)alloc((size_t)VOP * 352 * 2);
  _Float16* s3h    = (_Float16*)alloc((size_t)PTOT * 512 * 2);
  _Float16* v3l    = (_Float16*)alloc((size_t)PTOT * 3 * VOP * 2);
  _Float16* d3     = (_Float16*)alloc((size_t)PTOT * 3 * VOP * 2);
  _Float16* v3h    = (_Float16*)alloc((size_t)PTOT * 3 * VOP * 2);
  float*    smean3 = (float*)alloc((size_t)BATCH * 512 * 4);
  float*    vmean3 = (float*)alloc((size_t)BATCH * 3 * VOP * 4);
  _Float16* s4in   = (_Float16*)alloc((size_t)PTOT * 1024 * 2);
  _Float16* v4in   = (_Float16*)alloc((size_t)PTOT * 3 * 352 * 2);
  float*    vmean4 = (float*)alloc((size_t)BATCH * 3 * VOP * 4);
  // reuse buffers dead before conv_fuse outputs are produced
  _Float16* s4h = s3h;
  _Float16* v4l = v3l;
  _Float16* d4  = d3;
  _Float16* v4h = v3h;

  const int PB = PTOT / 256;

  knn_kernel<<<dim3(NPTS / 256, BATCH), 256, 0, stream>>>(x, idx);
  feat0_kernel<<<PB, 256, 0, stream>>>(x, idx, Wv2s, Ws0, bs0, Wv0, Wd0, s0, v0);
  block1_kernel<<<PB, 256, 0, stream>>>(s0, v0, Ws1, bs1, Wv1, Wd1, Wsf1, bsf1, Wvf1, Wdf1,
                                        s1, v1, sfw, vfw);
  mean_n_kernel<float><<<dim3(BATCH, 64), 256, 0, stream>>>(sfw, gsmean, 64, 1, 64, 64);
  mean_n_kernel<float><<<dim3(BATCH, 63), 256, 0, stream>>>(vfw, gvmean, 63, 1, 63, 63);
  fw2_kernel<<<BATCH, 64, 0, stream>>>(gsmean, gvmean, Wsf2, bsf2, Wvf2, Wdf2, gs2, gv2);
  block2_kernel<<<PB, 256, 0, stream>>>(s1, v1, gs2, gv2, Ws2, bs2, Wv2, Wd2, s2h, v2h);

  pack_w_kernel<<<(512 * 64 + 255) / 256, 256, 0, stream>>>(Ws3, Ws3h, 512, 64, 512, 64);
  pack_w_kernel<<<(VOP * 32 + 255) / 256, 256, 0, stream>>>(Wv3, Wv3h, 170, 21, VOP, 32);
  pack_w_kernel<<<(VOP * 32 + 255) / 256, 256, 0, stream>>>(Wd3, Wd3h, 170, 21, VOP, 32);
  pack_w_kernel<<<(512 * 1024 + 255) / 256, 256, 0, stream>>>(Ws4, Ws4h, 512, 1024, 512, 1024);
  pack_w_kernel<<<(VOP * 352 + 255) / 256, 256, 0, stream>>>(Wv4, Wv4h, 170, 340, VOP, 352);
  pack_w_kernel<<<(VOP * 352 + 255) / 256, 256, 0, stream>>>(Wd4, Wd4h, 170, 340, VOP, 352);

  // conv3 GEMMs: scalar M=32768 K=64 N=512 ; vector M=98304 K=32 N=192
  wmma_gemm_kernel<1><<<dim3(PTOT / GBM, 512 / GBN), 256, 0, stream>>>(s2h, Ws3h, bs3, s3h, 64, 512);
  wmma_gemm_kernel<0><<<dim3(PTOT * 3 / GBM, VOP / GBN), 256, 0, stream>>>(v2h, Wv3h, nullptr, v3l, 32, VOP);
  wmma_gemm_kernel<0><<<dim3(PTOT * 3 / GBM, VOP / GBN), 256, 0, stream>>>(v2h, Wd3h, nullptr, d3, 32, VOP);
  vn_leaky_kernel<<<(int)(((size_t)PTOT * VOP + 255) / 256), 256, 0, stream>>>(v3l, d3, v3h, 170, VOP);

  mean_n_kernel<_Float16><<<dim3(BATCH, 512), 256, 0, stream>>>(s3h, smean3, 512, 1, 512, 512);
  mean_n_kernel<_Float16><<<dim3(BATCH, 3 * VOP), 256, 0, stream>>>(v3h, vmean3, VOP, 3, VOP, VOP);
  pack_s4in_kernel<<<(int)(((size_t)PTOT * 1024 + 255) / 256), 256, 0, stream>>>(s3h, smean3, s4in);
  pack_v4in_kernel<<<(int)(((size_t)PTOT * 3 * 352 + 255) / 256), 256, 0, stream>>>(v3h, vmean3, v4in);

  // conv_fuse GEMMs: scalar M=32768 K=1024 N=512 ; vector M=98304 K=352 N=192
  wmma_gemm_kernel<1><<<dim3(PTOT / GBM, 512 / GBN), 256, 0, stream>>>(s4in, Ws4h, bs4, s4h, 1024, 512);
  wmma_gemm_kernel<0><<<dim3(PTOT * 3 / GBM, VOP / GBN), 256, 0, stream>>>(v4in, Wv4h, nullptr, v4l, 352, VOP);
  wmma_gemm_kernel<0><<<dim3(PTOT * 3 / GBM, VOP / GBN), 256, 0, stream>>>(v4in, Wd4h, nullptr, d4, 352, VOP);
  vn_leaky_kernel<<<(int)(((size_t)PTOT * VOP + 255) / 256), 256, 0, stream>>>(v4l, d4, v4h, 170, VOP);

  mean_n_kernel<_Float16><<<dim3(BATCH, 512), 256, 0, stream>>>(s4h, out, 512, 1, 512, 1022);
  mean_n_kernel<_Float16><<<dim3(BATCH, 3 * VOP), 256, 0, stream>>>(v4h, vmean4, VOP, 3, VOP, VOP);
  fuse_kernel<<<BATCH, 512, 0, stream>>>(vmean4, Wfuse, out);
}
